// KAE_29076928594330
// MI455X (gfx1250) — compile-verified, matrix operand order checked by reading, not run on hardware
//
#include <hip/hip_runtime.h>
#include <hip/hip_bf16.h>

typedef __bf16 v16bf __attribute__((ext_vector_type(16)));
typedef __bf16 v8bf  __attribute__((ext_vector_type(8)));
typedef __bf16 v4bf  __attribute__((ext_vector_type(4)));
typedef float  v8f   __attribute__((ext_vector_type(8)));
typedef float  v4f   __attribute__((ext_vector_type(4)));

#define KN 3072   // data_num

// ---------------------------------------------------------------------------
// f32 -> bf16 with zero column padding. Scalar fallback (used when C % 4 != 0).
// ---------------------------------------------------------------------------
__global__ void f32_to_bf16_pad(const float* __restrict__ in, __bf16* __restrict__ out,
                                int R, int C, int Cp) {
  size_t idx = (size_t)blockIdx.x * blockDim.x + threadIdx.x;
  size_t total = (size_t)R * Cp;
  if (idx >= total) return;
  int r = (int)(idx / Cp);
  int c = (int)(idx - (size_t)r * Cp);
  out[idx] = (c < C) ? (__bf16)in[(size_t)r * C + c] : (__bf16)0.0f;
}

// Vectorized variant: requires C % 4 == 0 and Cp % 4 == 0 (16B-aligned loads,
// 8B-aligned bf16 stores).
__global__ void f32_to_bf16_pad_v4(const float* __restrict__ in, __bf16* __restrict__ out,
                                   int R, int C, int Cp) {
  size_t vidx = (size_t)blockIdx.x * blockDim.x + threadIdx.x;
  size_t total = (size_t)R * Cp / 4;
  if (vidx >= total) return;
  size_t base = vidx * 4;
  int r = (int)(base / Cp);
  int c0 = (int)(base - (size_t)r * Cp);
  v4bf o;
  if (c0 + 3 < C) {
    v4f iv = *(const v4f*)(in + (size_t)r * C + c0);
#pragma unroll
    for (int t = 0; t < 4; ++t) o[t] = (__bf16)iv[t];
  } else {
#pragma unroll
    for (int t = 0; t < 4; ++t)
      o[t] = (c0 + t < C) ? (__bf16)in[(size_t)r * C + c0 + t] : (__bf16)0.0f;
  }
  *(v4bf*)(out + base) = o;
}

// f32 (R x C row-major) -> bf16 transpose, padded to Np x Kp (zeros outside).
// out[n*Kp + k] = in[k*C + n]   (inherently a gather; scalar is fine — weights
// are small and this runs once per GEMM).
__global__ void f32_to_bf16_pad_t(const float* __restrict__ in, __bf16* __restrict__ out,
                                  int R, int C, int Np, int Kp) {
  size_t idx = (size_t)blockIdx.x * blockDim.x + threadIdx.x;
  size_t total = (size_t)Np * Kp;
  if (idx >= total) return;
  int n = (int)(idx / Kp);
  int k = (int)(idx - (size_t)n * Kp);
  out[idx] = (n < C && k < R) ? (__bf16)in[(size_t)k * C + n] : (__bf16)0.0f;
}

// ---------------------------------------------------------------------------
// WMMA bf16 GEMM:  C(MxN, f32, ld=ldc) = A(MxKp) * Bt(NxKp)^T  (+ bias)
// K padded to 32 with zeros; Bt rows padded to a multiple of 128 so all
// fragment loads are unguarded 128-bit loads. Operands stay resident in the
// 192MB L2, so direct-global fragments hit L2 across waves.
// ---------------------------------------------------------------------------
__device__ __forceinline__ v16bf make_frag(const __bf16* __restrict__ p) {
  // CDNA5 16-bit A layout: lanes 0-15 row m, K = {8h+0..7, 16+8h+0..7}
  v8bf lo = *(const v8bf*)p;          // K = 8h .. 8h+7
  v8bf hi = *(const v8bf*)(p + 16);   // K = 16+8h .. 16+8h+7
  v16bf f;
#pragma unroll
  for (int i = 0; i < 8; ++i) { f[i] = lo[i]; f[i + 8] = hi[i]; }
  return f;
}

__device__ __forceinline__ void store_tile(v8f acc, int rbase, int col,
                                           int M, int Ncols,
                                           const float* __restrict__ bias,
                                           float* __restrict__ C, int ldc, int lh) {
  if (col >= Ncols) return;
  const float bv = bias ? bias[col] : 0.0f;
#pragma unroll
  for (int r = 0; r < 8; ++r) {
    const int row = rbase + (lh << 3) + r;   // C/D layout: VGPR r -> M = r + 8*lh
    if (row < M) C[(size_t)row * ldc + col] = acc[r] + bv;
  }
}

__global__ void __launch_bounds__(256)
wmma_gemm_bt(const __bf16* __restrict__ A, int lda,
             const __bf16* __restrict__ Bt, int ldb,
             const float* __restrict__ bias,
             float* __restrict__ C, int M, int Ncols, int Kp, int ldc) {
  const int lane = (int)(threadIdx.x & 31u);
  const int wave = (int)(threadIdx.x >> 5u);
  const int m0 = (int)blockIdx.y * 64 + (wave >> 2) * 32;   // 2 waves along M
  const int n0 = (int)blockIdx.x * 128 + (wave & 3) * 32;   // 4 waves along N

  v8f acc00 = {0.f, 0.f, 0.f, 0.f, 0.f, 0.f, 0.f, 0.f};
  v8f acc01 = acc00, acc10 = acc00, acc11 = acc00;

  const int mr = lane & 15, hh = lane >> 4;
  const __bf16* pa0 = A  + (size_t)(m0 + mr)      * lda + (hh << 3);
  const __bf16* pa1 = A  + (size_t)(m0 + 16 + mr) * lda + (hh << 3);
  const __bf16* pb0 = Bt + (size_t)(n0 + mr)      * ldb + (hh << 3);
  const __bf16* pb1 = Bt + (size_t)(n0 + 16 + mr) * ldb + (hh << 3);

  for (int kk = 0; kk < Kp; kk += 32) {
    __builtin_prefetch((const void*)(pa0 + 64), 0, 1);   // K tile two ahead of A
    __builtin_prefetch((const void*)(pb0 + 64), 0, 1);   // K tile two ahead of B
    v16bf a0 = make_frag(pa0); pa0 += 32;
    v16bf a1 = make_frag(pa1); pa1 += 32;
    v16bf b0 = make_frag(pb0); pb0 += 32;
    v16bf b1 = make_frag(pb1); pb1 += 32;
    acc00 = __builtin_amdgcn_wmma_f32_16x16x32_bf16(false, a0, false, b0, (short)0, acc00, false, false);
    acc01 = __builtin_amdgcn_wmma_f32_16x16x32_bf16(false, a0, false, b1, (short)0, acc01, false, false);
    acc10 = __builtin_amdgcn_wmma_f32_16x16x32_bf16(false, a1, false, b0, (short)0, acc10, false, false);
    acc11 = __builtin_amdgcn_wmma_f32_16x16x32_bf16(false, a1, false, b1, (short)0, acc11, false, false);
  }

  const int lc = lane & 15, lh = lane >> 4;
  store_tile(acc00, m0,      n0 + lc,      M, Ncols, bias, C, ldc, lh);
  store_tile(acc01, m0,      n0 + 16 + lc, M, Ncols, bias, C, ldc, lh);
  store_tile(acc10, m0 + 16, n0 + lc,      M, Ncols, bias, C, ldc, lh);
  store_tile(acc11, m0 + 16, n0 + 16 + lc, M, Ncols, bias, C, ldc, lh);
}

// ---------------------------------------------------------------------------
// Kernel-combine stages. s_i = |h_i|^2 == diag(G).
// ---------------------------------------------------------------------------
__global__ void diag_kernel(const float* __restrict__ G, float* __restrict__ s) {
  int i = (int)(blockIdx.x * blockDim.x + threadIdx.x);
  if (i < KN) s[i] = G[(size_t)i * KN + i];
}

// rowsum_i of A = exp(-0.5*sq) with diagonal removed (A_ii == 1 exactly since
// sq_ii == 0), then dinv = rsqrt(rowsum + 1e-12). One block per row, 128-bit
// row loads, LDS tree reduction.
__global__ void __launch_bounds__(256)
arow_kernel(const float* __restrict__ G, const float* __restrict__ s,
            float* __restrict__ dinv) {
  const int i = (int)blockIdx.x;
  const float* row = G + (size_t)i * KN;
  const float si = s[i];
  float sum = 0.f;
  for (int j0 = (int)threadIdx.x * 4; j0 < KN; j0 += 256 * 4) {
    v4f g4 = *(const v4f*)(row + j0);
    v4f s4 = *(const v4f*)(s + j0);
#pragma unroll
    for (int t = 0; t < 4; ++t) {
      float sq = fmaxf(si + s4[t] - 2.0f * g4[t], 0.0f);
      sum += __expf(-0.5f * sq);
    }
  }
  __shared__ float red[256];
  red[threadIdx.x] = sum;
  __syncthreads();
  for (int st = 128; st > 0; st >>= 1) {
    if ((int)threadIdx.x < st) red[threadIdx.x] += red[threadIdx.x + st];
    __syncthreads();
  }
  if (threadIdx.x == 0) dinv[i] = rsqrtf((red[0] - 1.0f) + 1e-12f);
}

// Fused, vectorized: read gram once, emit kh = e*lin + f*tk + g*pk + h*gk AND
// k (+)= coef*kh in the same pass. 8x b128 loads (9 when !first) + 2x b128 stores per thread.
__global__ void __launch_bounds__(256)
combine_kernel(const float* __restrict__ G,
               const float* __restrict__ pe, const float* __restrict__ pf,
               const float* __restrict__ pg, const float* __restrict__ ph,
               const float* __restrict__ pcoef,
               const float* __restrict__ s, const float* __restrict__ dinv,
               float* __restrict__ kh, float* __restrict__ kacc, int first) {
  size_t vidx = (size_t)blockIdx.x * blockDim.x + threadIdx.x;
  if (vidx >= (size_t)KN * KN / 4) return;
  const size_t base = vidx * 4;
  const int i  = (int)(base / KN);
  const int j0 = (int)(base - (size_t)i * KN);

  const v4f lin = *(const v4f*)(G + base);
  const v4f ve  = *(const v4f*)(pe + base);
  const v4f vf  = *(const v4f*)(pf + base);
  const v4f vg  = *(const v4f*)(pg + base);
  const v4f vh  = *(const v4f*)(ph + base);
  const v4f sj  = *(const v4f*)(s + j0);
  const v4f dj  = *(const v4f*)(dinv + j0);
  const float si = s[i];
  const float di = dinv[i];

  v4f v;
#pragma unroll
  for (int t = 0; t < 4; ++t) {
    const float sq  = fmaxf(si + sj[t] - 2.0f * lin[t], 0.0f);
    const float tk  = 1.0f / (1.0f + sq);
    const float lp1 = lin[t] + 1.0f;
    const float pk  = lp1 * lp1;
    const float gk  = (j0 + t == i) ? 0.0f : __expf(-0.5f * sq) * di * dj[t];
    v[t] = ve[t] * lin[t] + vf[t] * tk + vg[t] * pk + vh[t] * gk;
  }
  *(v4f*)(kh + base) = v;

  const v4f co = *(const v4f*)(pcoef + base);
  v4f kv = co * v;
  if (!first) kv += *(const v4f*)(kacc + base);
  *(v4f*)(kacc + base) = kv;
}

// q[i,c,j] = k[i,j] - cluster[c,j]  (vectorized along j; KN % 4 == 0)
__global__ void __launch_bounds__(256)
q_kernel(const float* __restrict__ k, const float* __restrict__ cl,
         float* __restrict__ q) {
  size_t vidx = (size_t)blockIdx.x * blockDim.x + threadIdx.x;
  if (vidx >= (size_t)KN * 10 * KN / 4) return;
  const size_t base = vidx * 4;
  const int j0 = (int)(base % KN);
  const size_t t = base / KN;
  const int c = (int)(t % 10);
  const int i = (int)(t / 10);
  const v4f kv = *(const v4f*)(k + (size_t)i * KN + j0);
  const v4f cv = *(const v4f*)(cl + (size_t)c * KN + j0);
  *(v4f*)(q + base) = kv - cv;
}

// ---------------------------------------------------------------------------
extern "C" void kernel_launch(void* const* d_in, const int* in_sizes, int n_in,
                              void* d_out, int out_size, void* d_ws, size_t ws_size,
                              hipStream_t stream) {
  (void)in_sizes; (void)n_in; (void)out_size; (void)ws_size;

  const float* x      = (const float*)d_in[0];
  const float* W_enc1 = (const float*)d_in[1];
  const float* b_enc1 = (const float*)d_in[2];
  const float* W_enc2 = (const float*)d_in[3];
  const float* b_enc2 = (const float*)d_in[4];
  const float* W_enc3 = (const float*)d_in[5];
  const float* b_enc3 = (const float*)d_in[6];
  const float* W_z    = (const float*)d_in[7];
  const float* b_z    = (const float*)d_in[8];
  const float* Wk1    = (const float*)d_in[9];
  const float* bk1    = (const float*)d_in[10];
  const float* Wk2    = (const float*)d_in[11];
  const float* bk2    = (const float*)d_in[12];
  const float* Wk3    = (const float*)d_in[13];
  const float* bk3    = (const float*)d_in[14];
  const float* Wk4    = (const float*)d_in[15];
  const float* bk4    = (const float*)d_in[16];
  const float* Pa  = (const float*)d_in[17];
  const float* Pb  = (const float*)d_in[18];
  const float* Pc  = (const float*)d_in[19];
  const float* Pd  = (const float*)d_in[20];
  const float* Pe  = (const float*)d_in[21];
  const float* Pf  = (const float*)d_in[22];
  const float* Pg  = (const float*)d_in[23];
  const float* Ph  = (const float*)d_in[24];
  const float* Pe1 = (const float*)d_in[25];
  const float* Pf1 = (const float*)d_in[26];
  const float* Pg1 = (const float*)d_in[27];
  const float* Ph1 = (const float*)d_in[28];
  const float* Pe2 = (const float*)d_in[29];
  const float* Pf2 = (const float*)d_in[30];
  const float* Pg2 = (const float*)d_in[31];
  const float* Ph2 = (const float*)d_in[32];
  const float* Pe3 = (const float*)d_in[33];
  const float* Pf3 = (const float*)d_in[34];
  const float* Pg3 = (const float*)d_in[35];
  const float* Ph3 = (const float*)d_in[36];
  const float* cluster = (const float*)d_in[37];

  // ---- d_out regions (return-order concatenation, all f32) ----
  float* out = (float*)d_out;
  constexpr size_t SZ_XBAR = (size_t)KN * 784;
  constexpr size_t SZ_K    = (size_t)KN * KN;
  constexpr size_t SZ_Z    = (size_t)KN * 10;
  constexpr size_t SZ_E    = (size_t)KN * 500;
  constexpr size_t SZ_E3   = (size_t)KN * 2000;
  constexpr size_t SZ_Q    = (size_t)KN * 10 * KN;
  float* o_xbar = out;
  float* o_k    = o_xbar + SZ_XBAR;
  float* o_z    = o_k    + SZ_K;
  float* o_e1   = o_z    + SZ_Z;
  float* o_e2   = o_e1   + SZ_E;
  float* o_e3   = o_e2   + SZ_E;
  float* o_q    = o_e3   + SZ_E3;
  float* o_k22  = o_q    + SZ_Q;
  float* o_k33  = o_k22  + SZ_E;
  float* o_kh1  = o_k33  + SZ_E;
  float* o_kh2  = o_kh1  + SZ_K;
  float* o_kh3  = o_kh2  + SZ_K;
  float* o_kz   = o_kh3  + SZ_K;

  // ---- workspace bump allocator ----
  char* wsb = (char*)d_ws;
  size_t off = 0;
  auto alloc = [&](size_t bytes) -> void* {
    off = (off + 255) & ~(size_t)255;
    void* p = wsb + off;
    off += bytes;
    return p;
  };
  float*  G     = (float*)alloc(sizeof(float) * (size_t)KN * KN);
  float*  k1f   = (float*)alloc(sizeof(float) * (size_t)KN * 2000);
  float*  sdiag = (float*)alloc(sizeof(float) * KN);
  float*  dinv  = (float*)alloc(sizeof(float) * KN);
  __bf16* xb    = (__bf16*)alloc(2ull * KN * 800);
  __bf16* W1t   = (__bf16*)alloc(2ull * 512 * 800);
  __bf16* enc1b = (__bf16*)alloc(2ull * KN * 512);
  __bf16* W2t   = (__bf16*)alloc(2ull * 512 * 512);
  __bf16* enc2b = (__bf16*)alloc(2ull * KN * 512);
  __bf16* W3t   = (__bf16*)alloc(2ull * 2048 * 512);
  __bf16* enc3b = (__bf16*)alloc(2ull * KN * 2016);
  __bf16* Wzt   = (__bf16*)alloc(2ull * 128 * 2016);
  __bf16* zb    = (__bf16*)alloc(2ull * KN * 32);
  __bf16* kb    = (__bf16*)alloc(2ull * KN * KN);
  __bf16* Wk1t  = (__bf16*)alloc(2ull * 2048 * KN);
  __bf16* k1b   = (__bf16*)alloc(2ull * KN * 2016);
  __bf16* Wk2t  = (__bf16*)alloc(2ull * 512 * 2016);
  __bf16* k22b  = (__bf16*)alloc(2ull * KN * 512);
  __bf16* Wk3t  = (__bf16*)alloc(2ull * 512 * 512);
  __bf16* k33b  = (__bf16*)alloc(2ull * KN * 512);
  __bf16* Wk4t  = (__bf16*)alloc(2ull * 896 * 512);

  // ---- launch helpers ----
  auto cvt = [&](const float* in, __bf16* ob, int R, int C, int Cp) {
    if ((C & 3) == 0) {
      size_t total = (size_t)R * Cp / 4;
      f32_to_bf16_pad_v4<<<dim3((unsigned)((total + 255) / 256)), dim3(256), 0, stream>>>(in, ob, R, C, Cp);
    } else {
      size_t total = (size_t)R * Cp;
      f32_to_bf16_pad<<<dim3((unsigned)((total + 255) / 256)), dim3(256), 0, stream>>>(in, ob, R, C, Cp);
    }
  };
  auto cvtT = [&](const float* in, __bf16* ob, int R, int C, int Np, int Kp) {
    size_t total = (size_t)Np * Kp;
    f32_to_bf16_pad_t<<<dim3((unsigned)((total + 255) / 256)), dim3(256), 0, stream>>>(in, ob, R, C, Np, Kp);
  };
  auto gemm = [&](const __bf16* A, int lda, const __bf16* Bt, int ldb,
                  const float* bias, float* C, int M, int Ncols, int Kp, int ldc) {
    dim3 grid((unsigned)((Ncols + 127) / 128), (unsigned)((M + 63) / 64));
    wmma_gemm_bt<<<grid, dim3(256), 0, stream>>>(A, lda, Bt, ldb, bias, C, M, Ncols, Kp, ldc);
  };
  auto stage = [&](const float* pe, const float* pf, const float* pg, const float* ph,
                   const float* pcoef, float* kh_out, int first) {
    diag_kernel<<<dim3(12), dim3(256), 0, stream>>>(G, sdiag);
    arow_kernel<<<dim3(KN), dim3(256), 0, stream>>>(G, sdiag, dinv);
    size_t total = (size_t)KN * KN / 4;
    combine_kernel<<<dim3((unsigned)((total + 255) / 256)), dim3(256), 0, stream>>>(
        G, pe, pf, pg, ph, pcoef, sdiag, dinv, kh_out, o_k, first);
  };

  // ---- stage 1 ----
  cvt(x, xb, KN, 784, 800);
  cvtT(W_enc1, W1t, 784, 500, 512, 800);
  gemm(xb, 800, W1t, 800, b_enc1, o_e1, KN, 500, 800, 500);
  cvt(o_e1, enc1b, KN, 500, 512);
  gemm(enc1b, 512, enc1b, 512, nullptr, G, KN, KN, 512, KN);
  stage(Pe1, Pf1, Pg1, Ph1, Pa, o_kh1, 1);

  // ---- stage 2 ----
  cvtT(W_enc2, W2t, 500, 500, 512, 512);
  gemm(enc1b, 512, W2t, 512, b_enc2, o_e2, KN, 500, 512, 500);
  cvt(o_e2, enc2b, KN, 500, 512);
  gemm(enc2b, 512, enc2b, 512, nullptr, G, KN, KN, 512, KN);
  stage(Pe2, Pf2, Pg2, Ph2, Pb, o_kh2, 0);

  // ---- stage 3 ----
  cvtT(W_enc3, W3t, 500, 2000, 2048, 512);
  gemm(enc2b, 512, W3t, 512, b_enc3, o_e3, KN, 2000, 512, 2000);
  cvt(o_e3, enc3b, KN, 2000, 2016);
  gemm(enc3b, 2016, enc3b, 2016, nullptr, G, KN, KN, 2016, KN);
  stage(Pe3, Pf3, Pg3, Ph3, Pc, o_kh3, 0);

  // ---- z stage ----
  cvtT(W_z, Wzt, 2000, 10, 128, 2016);
  gemm(enc3b, 2016, Wzt, 2016, b_z, o_z, KN, 10, 2016, 10);
  cvt(o_z, zb, KN, 10, 32);                 // C=10 -> scalar fallback path
  gemm(zb, 32, zb, 32, nullptr, G, KN, KN, 32, KN);
  stage(Pe, Pf, Pg, Ph, Pd, o_kz, 0);

  // ---- decoder ----
  cvt(o_k, kb, KN, KN, KN);
  cvtT(Wk1, Wk1t, KN, 2000, 2048, KN);
  gemm(kb, KN, Wk1t, KN, bk1, k1f, KN, 2000, KN, 2000);
  cvt(k1f, k1b, KN, 2000, 2016);
  cvtT(Wk2, Wk2t, 2000, 500, 512, 2016);
  gemm(k1b, 2016, Wk2t, 2016, bk2, o_k22, KN, 500, 2016, 500);
  cvt(o_k22, k22b, KN, 500, 512);
  cvtT(Wk3, Wk3t, 500, 500, 512, 512);
  gemm(k22b, 512, Wk3t, 512, bk3, o_k33, KN, 500, 512, 500);
  cvt(o_k33, k33b, KN, 500, 512);
  cvtT(Wk4, Wk4t, 500, 784, 896, 512);
  gemm(k33b, 512, Wk4t, 512, bk4, o_xbar, KN, 784, 512, 784);

  // ---- q ----
  size_t qtot = (size_t)KN * 10 * KN / 4;
  q_kernel<<<dim3((unsigned)((qtot + 255) / 256)), dim3(256), 0, stream>>>(o_k, cluster, o_q);
}
